// EncoderLayer_30562987278917
// MI455X (gfx1250) — matrix operand order repaired and checked by reading, not compiled, not run on hardware
//
#include <hip/hip_runtime.h>
#include <hip/hip_bf16.h>

// ---------------------------------------------------------------------------
// Mamba2 encoder layer for MI455X (gfx1250, wave32).
//  * GEMM operands held in bf16 in HBM (producers convert) -> halves GEMM
//    read traffic vs fp32 (23.3 TB/s HBM roofline).
//  * GEMM K-loop is a double-buffered async pipeline: tile k+1 staged with
//    GLOBAL_LOAD_ASYNC_TO_LDS_B128 (ASYNCcnt) while 4x v_wmma_f32_16x16x32_bf16
//    consume tile k; one s_wait_asynccnt + one barrier per K-step.
//  * Fragments are contiguous 16B LDS runs -> ds_load_b128 pairs (ISA 7.12.2).
//  * SSM scan: per-(batch,head) block recurrence, state in registers.
// ---------------------------------------------------------------------------

#define BB    2
#define LL    4096
#define DM    1024
#define STATE 128
#define DIN   2048            // D_INNER
#define NH    32              // heads
#define HD    64              // head dim
#define CDIM  2304            // CONV_DIM = DIN + 2*STATE
#define DIP   4384            // D_IN_PROJ
#define MLPI  4096
#define MROWS (BB * LL)       // 8192

typedef __attribute__((ext_vector_type(16))) __bf16 v16bf;
typedef __attribute__((ext_vector_type(8)))  __bf16 v8bf;
typedef __attribute__((ext_vector_type(8)))  float  v8f;
typedef __attribute__((ext_vector_type(4)))  int    v4i;

#if __has_builtin(__builtin_amdgcn_global_load_async_to_lds_b128) && \
    __has_builtin(__builtin_amdgcn_s_wait_asynccnt)
#define HAVE_ASYNC_LDS 1
#else
#define HAVE_ASYNC_LDS 0
#endif

__device__ __forceinline__ __bf16 f2bf(float f) {
    unsigned u = __builtin_bit_cast(unsigned, f);
    unsigned short s = (unsigned short)((u + 0x7FFFu + ((u >> 16) & 1u)) >> 16); // RNE
    return __builtin_bit_cast(__bf16, s);
}
__device__ __forceinline__ float silu_f(float x) { return x / (1.f + __expf(-x)); }

// 16-byte global -> LDS chunk copy (async path when available).
// Builtin signature (probe-verified): (v4i AS1* src, v4i AS3* dst, Ii, Ii)
__device__ __forceinline__ void copy16_g2l(const __bf16* g, __bf16* l) {
#if HAVE_ASYNC_LDS
    __builtin_amdgcn_global_load_async_to_lds_b128(
        (__attribute__((address_space(1))) v4i*)g,
        (__attribute__((address_space(3))) v4i*)l, 0, 0);
#else
    *(v8bf*)l = *(const v8bf*)g;
#endif
}

// ---------------------------------------------------------------------------
// GEMM: Y[M,N] = X[M,K] @ W[N,K]^T  (bf16 in, fp32 out)
// Block tile 128x64, 8 waves as 4(M)x2(N); each wave a 32x32 tile = 2x2 WMMA.
// K multiples of 32 required; ragged N (4384) via clamped loads/guarded stores.
// ---------------------------------------------------------------------------
#define GBM  128
#define GBN  64
#define GBK  32
#define LDAB 40    // LDS row stride in bf16 (80 B: 16B-aligned chunks, bank skew)

__global__ __launch_bounds__(256)
void gemm_bf16_wmma(const __bf16* __restrict__ X, const __bf16* __restrict__ W,
                    float* __restrict__ Y, int M, int N, int K) {
    __shared__ __attribute__((aligned(32))) __bf16 As[2][GBM * LDAB];
    __shared__ __attribute__((aligned(32))) __bf16 Bs[2][GBN * LDAB];

    const int tid    = threadIdx.x;
    const int m0     = blockIdx.y * GBM;
    const int n0     = blockIdx.x * GBN;
    const int w      = tid >> 5;
    const int lane   = tid & 31;
    const int laneHi = lane >> 4;
    const int lane15 = lane & 15;
    const int wm     = (w & 3) * 32;   // wave M offset in block
    const int wn     = (w >> 2) * 32;  // wave N offset in block

    // Stage one 128x32 (A) + 64x32 (B) bf16 tile pair into LDS buffer `buf`.
    auto stage = [&](int buf, int k0) {
        #pragma unroll
        for (int i = 0; i < 2; ++i) {                 // A: 512 chunks, 2/thread
            const int chunk = tid * 2 + i;
            const int row   = chunk >> 2;
            const int cpos  = (chunk & 3) * 8;
            copy16_g2l(X + (size_t)(m0 + row) * K + k0 + cpos,
                       &As[buf][row * LDAB + cpos]);
        }
        {                                             // B: 256 chunks, 1/thread
            const int row  = tid >> 2;
            const int gr   = (n0 + row < N) ? (n0 + row) : (N - 1);
            const int cpos = (tid & 3) * 8;
            copy16_g2l(W + (size_t)gr * K + k0 + cpos,
                       &Bs[buf][row * LDAB + cpos]);
        }
    };

    v8f acc[2][2] = {};
    stage(0, 0);                       // prologue: fill buffer 0
    int cur = 0;

    for (int k0 = 0; k0 < K; k0 += GBK) {
#if HAVE_ASYNC_LDS
        __builtin_amdgcn_s_wait_asynccnt(0);   // this wave's staged chunks done
#endif
        __syncthreads();                        // publish buf[cur] to all waves

        if (k0 + GBK < K)                       // overlap: prefetch next tile
            stage(cur ^ 1, k0 + GBK);

        // ---- Fragments (ISA 7.12.2). A lane needs K {0-7,16-23}+8*laneHi,
        //      B lane needs the same K {0-7,16-23}+8*laneHi: all contiguous 16B LDS runs.
        v16bf af[2], bfv[2];
        #pragma unroll
        for (int s = 0; s < 2; ++s) {
            const __bf16* ar = &As[cur][(wm + s * 16 + lane15) * LDAB];
            const v8bf alo = *(const v8bf*)(ar + laneHi * 8);
            const v8bf ahi = *(const v8bf*)(ar + 16 + laneHi * 8);
            const __bf16* br = &Bs[cur][(wn + s * 16 + lane15) * LDAB + laneHi * 8];
            const v8bf blo = *(const v8bf*)br;
            const v8bf bhi = *(const v8bf*)(br + 16);
            #pragma unroll
            for (int e = 0; e < 8; ++e) {
                af[s][e]  = alo[e];  af[s][8 + e]  = ahi[e];
                bfv[s][e] = blo[e];  bfv[s][8 + e] = bhi[e];
            }
        }
        #pragma unroll
        for (int i = 0; i < 2; ++i)
            #pragma unroll
            for (int j = 0; j < 2; ++j)
                acc[i][j] = __builtin_amdgcn_wmma_f32_16x16x32_bf16(
                                false, af[i], false, bfv[j],
                                (short)0, acc[i][j], false, false);
        cur ^= 1;
    }

    // C/D layout: VGPR r -> M = r + 8*laneHi, N = lane&15 (per 16x16 subtile)
    #pragma unroll
    for (int j = 0; j < 2; ++j) {
        const int col = n0 + wn + j * 16 + lane15;
        if (col < N) {
            #pragma unroll
            for (int i = 0; i < 2; ++i)
                #pragma unroll
                for (int r = 0; r < 8; ++r) {
                    const int row = m0 + wm + i * 16 + r + laneHi * 8;
                    Y[(size_t)row * N + col] = acc[i][j][r];
                }
        }
    }
}

// ---------------------------------------------------------------------------
// One-shot fp32 -> bf16 conversion (weights). n must be a multiple of 1024.
// ---------------------------------------------------------------------------
__global__ __launch_bounds__(256)
void f32_to_bf16_kernel(const float* __restrict__ src, __bf16* __restrict__ dst) {
    const size_t i = ((size_t)blockIdx.x * 256 + threadIdx.x) * 4;
    #pragma unroll
    for (int j = 0; j < 4; ++j) dst[i + j] = f2bf(src[i + j]);
}

// ---------------------------------------------------------------------------
// RMSNorm prenorm: res = x + res_in (f32); normed (bf16) = res * rsqrt(...) * w
// ---------------------------------------------------------------------------
__global__ __launch_bounds__(256)
void rmsnorm_prenorm_kernel(const float* __restrict__ x, const float* __restrict__ res_in,
                            const float* __restrict__ w, __bf16* __restrict__ normed,
                            float* __restrict__ res_out) {
    __shared__ float red[8];
    const size_t row = blockIdx.x;
    const float* xr = x      + row * DM;
    const float* rr = res_in + row * DM;
    float vals[4];
    float ss = 0.f;
    #pragma unroll
    for (int j = 0; j < 4; ++j) {
        const int i = threadIdx.x + j * 256;
        const float v = xr[i] + rr[i];
        vals[j] = v;
        res_out[row * DM + i] = v;
        ss += v * v;
    }
    #pragma unroll
    for (int o = 16; o > 0; o >>= 1) ss += __shfl_xor(ss, o, 32);
    if ((threadIdx.x & 31) == 0) red[threadIdx.x >> 5] = ss;
    __syncthreads();
    float tot = 0.f;
    #pragma unroll
    for (int i = 0; i < 8; ++i) tot += red[i];
    const float rs = rsqrtf(tot / (float)DM + 1e-5f);
    #pragma unroll
    for (int j = 0; j < 4; ++j) {
        const int i = threadIdx.x + j * 256;
        normed[row * DM + i] = f2bf(vals[j] * rs * w[i]);
    }
}

// ---------------------------------------------------------------------------
// Causal depthwise conv1d (taps=4) + bias + silu over xBC channels of zxbcdt.
// ---------------------------------------------------------------------------
__global__ __launch_bounds__(256)
void conv_silu_kernel(const float* __restrict__ zxb, const float* __restrict__ cw,
                      const float* __restrict__ cb, float* __restrict__ xBC) {
    const size_t idx = (size_t)blockIdx.x * 256 + threadIdx.x;
    if (idx >= (size_t)MROWS * CDIM) return;
    const int    c  = (int)(idx % CDIM);
    const size_t bl = idx / CDIM;
    const int    l  = (int)(bl % LL);
    const size_t b  = bl / LL;
    float acc = cb[c];
    #pragma unroll
    for (int j = 0; j < 4; ++j) {
        const int lj = l - 3 + j;
        if (lj >= 0)
            acc += zxb[(b * LL + lj) * (size_t)DIP + DIN + c] * cw[c * 4 + j];
    }
    xBC[idx] = silu_f(acc);
}

// ---------------------------------------------------------------------------
// SSM selective scan. grid = B*NH blocks; thread (p = tid>>2, q = tid&3) owns
// state[p, q*32 .. q*32+31] in registers; wave32 shfl_xor reduces n.
// ---------------------------------------------------------------------------
__global__ __launch_bounds__(256)
void ssm_scan_kernel(const float* __restrict__ zxb, const float* __restrict__ xBC,
                     const float* __restrict__ dt_bias, const float* __restrict__ A_log,
                     float* __restrict__ ycore) {
    const int b   = blockIdx.x >> 5;
    const int h   = blockIdx.x & 31;
    const int tid = threadIdx.x;
    const int p   = tid >> 2;
    const int q   = tid & 3;
    const int n0  = q * 32;

    const float A   = -__expf(A_log[h]);
    const float dtb = dt_bias[h];

    __shared__ float sX[HD], sB[STATE], sC[STATE], sdt[1];

    float st[32];
    #pragma unroll
    for (int i = 0; i < 32; ++i) st[i] = 0.f;

    for (int t = 0; t < LL; ++t) {
        const size_t rowX = (size_t)(b * LL + t) * CDIM;
        if (tid < 128)       sB[tid]        = xBC[rowX + DIN + tid];
        else                 sC[tid - 128]  = xBC[rowX + DIN + STATE + (tid - 128)];
        if (tid < HD)        sX[tid]        = xBC[rowX + h * HD + tid];
        if (tid == 255) {
            const float draw = zxb[(size_t)(b * LL + t) * DIP + DIN + CDIM + h] + dtb;
            sdt[0] = (draw > 20.f) ? draw : __logf(1.f + __expf(draw));   // softplus
        }
        if (t + 1 < LL && tid == 0)
            __builtin_prefetch(xBC + rowX + CDIM, 0, 0);                  // global_prefetch
        __syncthreads();

        const float dt   = sdt[0];
        const float dA   = __expf(dt * A);
        const float coef = dt * sX[p];
        float acc = 0.f;
        #pragma unroll
        for (int i = 0; i < 32; ++i) {
            const float s2 = st[i] * dA + coef * sB[n0 + i];
            st[i] = s2;
            acc  += s2 * sC[n0 + i];
        }
        acc += __shfl_xor(acc, 1, 32);
        acc += __shfl_xor(acc, 2, 32);
        if (q == 0)
            ycore[(size_t)(b * LL + t) * DIN + h * HD + p] = acc;
        __syncthreads();
    }
}

// y (bf16) = (ycore + D[h]*xh) * silu(z)
__global__ __launch_bounds__(256)
void gate_mul_kernel(const float* __restrict__ ycore, const float* __restrict__ xBC,
                     const float* __restrict__ zxb, const float* __restrict__ Dp,
                     __bf16* __restrict__ y) {
    const size_t idx = (size_t)blockIdx.x * 256 + threadIdx.x;
    if (idx >= (size_t)MROWS * DIN) return;
    const int    c   = (int)(idx % DIN);
    const size_t row = idx / DIN;
    const int    h   = c >> 6;
    const float xh = xBC[row * CDIM + c];
    const float z  = zxb[row * (size_t)DIP + c];
    y[idx] = f2bf((ycore[idx] + Dp[h] * xh) * silu_f(z));
}

// out (bf16) = silu(g) * u
__global__ __launch_bounds__(256)
void silu_mul_kernel(const float* __restrict__ g, const float* __restrict__ u,
                     __bf16* __restrict__ out, size_t n) {
    const size_t idx = (size_t)blockIdx.x * 256 + threadIdx.x;
    if (idx >= n) return;
    out[idx] = f2bf(silu_f(g[idx]) * u[idx]);
}

// ---------------------------------------------------------------------------
extern "C" void kernel_launch(void* const* d_in, const int* in_sizes, int n_in,
                              void* d_out, int out_size, void* d_ws, size_t ws_size,
                              hipStream_t stream) {
    const float* hidden   = (const float*)d_in[0];
    const float* residual = (const float*)d_in[1];
    const float* ssm_nw   = (const float*)d_in[2];
    const float* mlp_nw   = (const float*)d_in[3];
    const float* in_proj  = (const float*)d_in[4];
    const float* conv_w   = (const float*)d_in[5];
    const float* conv_b   = (const float*)d_in[6];
    const float* dt_bias  = (const float*)d_in[7];
    const float* A_log    = (const float*)d_in[8];
    const float* D_param  = (const float*)d_in[9];
    const float* out_proj = (const float*)d_in[10];
    const float* gate_w   = (const float*)d_in[11];
    const float* up_w     = (const float*)d_in[12];
    const float* down_w   = (const float*)d_in[13];

    float* out_h   = (float*)d_out;                      // (B,L,DM)
    float* out_res = (float*)d_out + (size_t)MROWS * DM; // (B,L,DM)

    // ---- Workspace carve (assumes generous ws_size; ~800 MB total) ----
    char* p = (char*)d_ws;
    auto alloc = [&p](size_t bytes) {
        char* r = p;
        p += (bytes + 255) & ~(size_t)255;
        return r;
    };
    float*  res1   = (float*)alloc((size_t)MROWS * DM   * 4);
    float*  zxb    = (float*)alloc((size_t)MROWS * DIP  * 4);
    float*  xbc    = (float*)alloc((size_t)MROWS * CDIM * 4);
    float*  ycore  = (float*)alloc((size_t)MROWS * DIN  * 4);
    float*  mamba  = (float*)alloc((size_t)MROWS * DM   * 4);
    float*  gbuf   = (float*)alloc((size_t)MROWS * MLPI * 4);
    float*  ubuf   = (float*)alloc((size_t)MROWS * MLPI * 4);
    __bf16* h1b    = (__bf16*)alloc((size_t)MROWS * DM   * 2);
    __bf16* ybufb  = (__bf16*)alloc((size_t)MROWS * DIN  * 2);
    __bf16* h2b    = (__bf16*)alloc((size_t)MROWS * DM   * 2);
    __bf16* abuf   = (__bf16*)alloc((size_t)MROWS * MLPI * 2);
    __bf16* w_in   = (__bf16*)alloc((size_t)DIP  * DM   * 2);
    __bf16* w_out  = (__bf16*)alloc((size_t)DM   * DIN  * 2);
    __bf16* w_gate = (__bf16*)alloc((size_t)MLPI * DM   * 2);
    __bf16* w_up   = (__bf16*)alloc((size_t)MLPI * DM   * 2);
    __bf16* w_down = (__bf16*)alloc((size_t)DM   * MLPI * 2);

    const unsigned gM = MROWS / GBM;   // 64

    // 0) one-shot weight conversion to bf16 (all sizes are multiples of 1024)
    f32_to_bf16_kernel<<<(unsigned)((size_t)DIP  * DM   / 1024), 256, 0, stream>>>(in_proj,  w_in);
    f32_to_bf16_kernel<<<(unsigned)((size_t)DM   * DIN  / 1024), 256, 0, stream>>>(out_proj, w_out);
    f32_to_bf16_kernel<<<(unsigned)((size_t)MLPI * DM   / 1024), 256, 0, stream>>>(gate_w,   w_gate);
    f32_to_bf16_kernel<<<(unsigned)((size_t)MLPI * DM   / 1024), 256, 0, stream>>>(up_w,     w_up);
    f32_to_bf16_kernel<<<(unsigned)((size_t)DM   * MLPI / 1024), 256, 0, stream>>>(down_w,   w_down);

    // 1) prenorm 1 -> h1b (bf16), res1 (f32)
    rmsnorm_prenorm_kernel<<<MROWS, 256, 0, stream>>>(hidden, residual, ssm_nw, h1b, res1);

    // 2) in_proj: (M,1024) @ (4384,1024)^T
    gemm_bf16_wmma<<<dim3((DIP + GBN - 1) / GBN, gM), 256, 0, stream>>>(h1b, w_in, zxb, MROWS, DIP, DM);

    // 3) conv + silu over xBC channels
    {
        const size_t n = (size_t)MROWS * CDIM;
        conv_silu_kernel<<<(unsigned)((n + 255) / 256), 256, 0, stream>>>(zxb, conv_w, conv_b, xbc);
    }

    // 4) selective scan
    ssm_scan_kernel<<<BB * NH, 256, 0, stream>>>(zxb, xbc, dt_bias, A_log, ycore);

    // 5) y = (ycore + D*xh) * silu(z) -> bf16
    {
        const size_t n = (size_t)MROWS * DIN;
        gate_mul_kernel<<<(unsigned)((n + 255) / 256), 256, 0, stream>>>(ycore, xbc, zxb, D_param, ybufb);
    }

    // 6) out_proj: (M,2048) @ (1024,2048)^T
    gemm_bf16_wmma<<<dim3(DM / GBN, gM), 256, 0, stream>>>(ybufb, w_out, mamba, MROWS, DM, DIN);

    // 7) prenorm 2 -> h2b (bf16), res2 straight to output
    rmsnorm_prenorm_kernel<<<MROWS, 256, 0, stream>>>(mamba, res1, mlp_nw, h2b, out_res);

    // 8) MLP gate / up GEMMs
    gemm_bf16_wmma<<<dim3(MLPI / GBN, gM), 256, 0, stream>>>(h2b, w_gate, gbuf, MROWS, MLPI, DM);
    gemm_bf16_wmma<<<dim3(MLPI / GBN, gM), 256, 0, stream>>>(h2b, w_up,   ubuf, MROWS, MLPI, DM);

    // 9) silu(g) * u -> bf16
    {
        const size_t n = (size_t)MROWS * MLPI;
        silu_mul_kernel<<<(unsigned)((n + 255) / 256), 256, 0, stream>>>(gbuf, ubuf, abuf, n);
    }

    // 10) down: (M,4096) @ (1024,4096)^T -> final hidden output
    gemm_bf16_wmma<<<dim3(DM / GBN, gM), 256, 0, stream>>>(abuf, w_down, out_h, MROWS, DM, MLPI);
}
